// PolicyNetwork_47012712022391
// MI455X (gfx1250) — matrix operand order repaired and checked by reading, not compiled, not run on hardware
//
#include <hip/hip_runtime.h>
#include <hip/hip_bf16.h>
#include <math.h>

#define NNODES 100000
#define NEDGES 1600000
#define HID 128
#define MAXNN 100000
#define TAU 0.5f
#define MASK_VALUE -1000000000.0f

typedef __attribute__((ext_vector_type(16))) _Float16 v16h;
typedef __attribute__((ext_vector_type(8)))  _Float16 v8h;
typedef __attribute__((ext_vector_type(8)))  float    v8f;
typedef __attribute__((ext_vector_type(4)))  int      v4i;

// ---------------------------------------------------------------- async global->LDS copy
// gfx1250 async path (ASYNCcnt-tracked); falls back to load+ds_store if the
// builtin is not available on this toolchain. Probe-learned prototype:
//   void __builtin_amdgcn_global_load_async_to_lds_b128(v4i AS1*, v4i AS3*, Imm, Imm)
#if __has_builtin(__builtin_amdgcn_global_load_async_to_lds_b128)
#define HAVE_ASYNC_LDS 1
#else
#define HAVE_ASYNC_LDS 0
#endif

__device__ __forceinline__ void lds_copy_b128(const _Float16* g, _Float16* l) {
#if HAVE_ASYNC_LDS
  _Float16* gnc = const_cast<_Float16*>(g);
  __builtin_amdgcn_global_load_async_to_lds_b128(
      (__attribute__((address_space(1))) v4i*)gnc,
      (__attribute__((address_space(3))) v4i*)l, 0, 0);
#else
  *(v8h*)l = *(const v8h*)g;
#endif
}
__device__ __forceinline__ void wait_async_copies() {
#if HAVE_ASYNC_LDS
#if __has_builtin(__builtin_amdgcn_s_wait_asynccnt)
  __builtin_amdgcn_s_wait_asynccnt(0);
#else
  asm volatile("s_wait_asynccnt 0x0" ::: "memory");
#endif
#endif
}

// ---------------------------------------------------------------- utilities
__global__ void k_zero_f32(float* p, int n) {
  for (int i = blockIdx.x * blockDim.x + threadIdx.x; i < n; i += gridDim.x * blockDim.x)
    p[i] = 0.0f;
}
__global__ void k_zero_i32(int* p, int n) {
  for (int i = blockIdx.x * blockDim.x + threadIdx.x; i < n; i += gridDim.x * blockDim.x)
    p[i] = 0;
}
__global__ void k_f32_to_f16(const float* __restrict__ a, _Float16* __restrict__ o, int n) {
  for (int i = blockIdx.x * blockDim.x + threadIdx.x; i < n; i += gridDim.x * blockDim.x)
    o[i] = (_Float16)a[i];
}
// W[128,128] row-major -> Wt[n][k] = W[k][n] in f16 (B-fragment K-runs contiguous)
__global__ void k_w_transpose_f16(const float* __restrict__ W, _Float16* __restrict__ Wt) {
  int i = blockIdx.x * blockDim.x + threadIdx.x;
  if (i < HID * HID) {
    int nn = i >> 7, k = i & 127;
    Wt[i] = (_Float16)W[k * HID + nn];
  }
}

// -------------------------------------------------------- degree / norm
__global__ void k_deg(const int* __restrict__ dst, int e, float* deg) {
  for (int i = blockIdx.x * blockDim.x + threadIdx.x; i < e; i += gridDim.x * blockDim.x)
    atomicAdd(&deg[dst[i]], 1.0f);
}
__global__ void k_dinv(float* deg, int n) {
  for (int i = blockIdx.x * blockDim.x + threadIdx.x; i < n; i += gridDim.x * blockDim.x)
    deg[i] = rsqrtf(deg[i] + 1.0f);     // +1 self loop
}

// -------------------------------------------------------- WMMA GEMM (fused self-loop init)
//   H = A @ W              (needed by the edge gather)
//   O = bias + H * dinv^2  (scatter-add target, self-loop + bias pre-applied)
// 32 rows per block (100000 % 32 == 0 -> no guards). A-tile (8KB) and W (32KB)
// staged in LDS via async copies; 8 waves each compute two 16x16 tiles.
__device__ __forceinline__ v16h pack16(const _Float16* lo, const _Float16* hi2) {
  v8h a = *(const v8h*)lo;
  v8h b = *(const v8h*)hi2;
  v16h r;
#pragma unroll
  for (int j = 0; j < 8; ++j) { r[j] = a[j]; r[8 + j] = b[j]; }
  return r;
}

__global__ __launch_bounds__(256) void k_gemm_f16(const _Float16* __restrict__ A,
                                                  const _Float16* __restrict__ Bt,
                                                  const float* __restrict__ bias,
                                                  const float* __restrict__ dinv,
                                                  float* __restrict__ H,
                                                  float* __restrict__ O) {
  __shared__ _Float16 sA[32 * HID];      // 8 KB
  __shared__ _Float16 sB[HID * HID];     // 32 KB
  const int tid = threadIdx.x;
  const int m0  = blockIdx.x * 32;

  // ---- stage A (512 x b128) and B (2048 x b128) into LDS
  const _Float16* gA = A + (size_t)m0 * HID;
#pragma unroll
  for (int i = 0; i < 2; ++i) {
    int c = tid + i * 256;
    lds_copy_b128(gA + c * 8, &sA[c * 8]);
  }
#pragma unroll
  for (int i = 0; i < 8; ++i) {
    int c = tid + i * 256;
    lds_copy_b128(Bt + c * 8, &sB[c * 8]);
  }
  wait_async_copies();
  __syncthreads();

  const int wave = tid >> 5;
  const int lane = tid & 31;
  const int ml   = lane & 15;
  const int hi   = lane >> 4;
  const int n0   = wave * 16;

  const _Float16* sBcol  = &sB[(n0 + ml) * HID];
  const _Float16* sArow0 = &sA[(ml)      * HID];
  const _Float16* sArow1 = &sA[(16 + ml) * HID];

  v8f acc0 = {}, acc1 = {};
#pragma unroll
  for (int k0 = 0; k0 < HID; k0 += 32) {
    // B fragment (32x16): lane half 'hi' holds K = {hi*8..+7, 16+hi*8..+7} of column n
    v16h b = pack16(sBcol + k0 + hi * 8, sBcol + k0 + 16 + hi * 8);
    // A fragment (16x32): lane half 'hi' holds K = {hi*8..+7, 16+hi*8..+7}
    v16h a0 = pack16(sArow0 + k0 + hi * 8, sArow0 + k0 + 16 + hi * 8);
    acc0 = __builtin_amdgcn_wmma_f32_16x16x32_f16(false, a0, false, b, (short)0, acc0,
                                                  false, false);
    v16h a1 = pack16(sArow1 + k0 + hi * 8, sArow1 + k0 + 16 + hi * 8);
    acc1 = __builtin_amdgcn_wmma_f32_16x16x32_f16(false, a1, false, b, (short)0, acc1,
                                                  false, false);
  }

  // ---- stores: D layout lane(hi,ml): col = n0+ml, VGPR r -> row m0 + hi*8 + r (+16)
  const int col = n0 + ml;
  const float bcol = bias[col];
#pragma unroll
  for (int r = 0; r < 8; ++r) {
    int rr = m0 + hi * 8 + r;
    float d = dinv[rr];
    size_t idx = (size_t)rr * HID + col;
    H[idx] = acc0[r];
    O[idx] = bcol + acc0[r] * d * d;
  }
#pragma unroll
  for (int r = 0; r < 8; ++r) {
    int rr = m0 + 16 + hi * 8 + r;
    float d = dinv[rr];
    size_t idx = (size_t)rr * HID + col;
    H[idx] = acc1[r];
    O[idx] = bcol + acc1[r] * d * d;
  }
}

// -------------------------------------------------------- edge scatter
// one wave per edge: gather 512B row, 4 float atomics per lane
__global__ __launch_bounds__(256) void k_edge_scatter(const int* __restrict__ src,
                                                      const int* __restrict__ dst,
                                                      const float* __restrict__ h,
                                                      const float* __restrict__ dinv,
                                                      float* __restrict__ out, int e) {
  int eIdx = blockIdx.x * (blockDim.x >> 5) + (threadIdx.x >> 5);
  if (eIdx >= e) return;
  int lane = threadIdx.x & 31;
  int s = src[eIdx], d = dst[eIdx];
  float scale = dinv[s] * dinv[d];
  float4 v = ((const float4*)(h + (size_t)s * HID))[lane];
  float* op = out + (size_t)d * HID + lane * 4;
  atomicAdd(op + 0, v.x * scale);
  atomicAdd(op + 1, v.y * scale);
  atomicAdd(op + 2, v.z * scale);
  atomicAdd(op + 3, v.w * scale);
}
__global__ void k_relu_cvt(float* __restrict__ out, _Float16* __restrict__ actH, int total) {
  for (int i = blockIdx.x * blockDim.x + threadIdx.x; i < total; i += gridDim.x * blockDim.x) {
    float v = out[i];
    v = v > 0.0f ? v : 0.0f;
    out[i]  = v;
    actH[i] = (_Float16)v;
  }
}

// -------------------------------------------------------- pooling (mean | max)
__global__ __launch_bounds__(128) void k_pool_stage1(const float* __restrict__ g, int n,
                                                     float* psum, float* pmax) {
  int c = threadIdx.x;
  int nb = gridDim.x;
  int chunk = (n + nb - 1) / nb;
  int r0 = blockIdx.x * chunk;
  int r1 = min(n, r0 + chunk);
  float s = 0.0f, m = -INFINITY;
  for (int r = r0; r < r1; ++r) {
    float v = g[(size_t)r * HID + c];
    s += v;
    m = fmaxf(m, v);
  }
  psum[blockIdx.x * HID + c] = s;
  pmax[blockIdx.x * HID + c] = m;
}
__global__ __launch_bounds__(128) void k_pool_stage2(const float* psum, const float* pmax,
                                                     int nb, int n, float* pool) {
  int c = threadIdx.x;
  float s = 0.0f, m = -INFINITY;
  for (int b = 0; b < nb; ++b) {
    s += psum[b * HID + c];
    m = fmaxf(m, pmax[b * HID + c]);
  }
  pool[c]       = s / (float)n;
  pool[HID + c] = m;
}

// -------------------------------------------------------- head GEMV + gumbel logits
// z[j] = (mask(pool . W[:,j] + b[j]) + g[j]) / TAU    (coalesced column reads)
__global__ __launch_bounds__(256) void k_head_gemv(const float* __restrict__ pool,
                                                   const float* __restrict__ W,
                                                   const float* __restrict__ b,
                                                   const float* __restrict__ g,
                                                   const int* __restrict__ conn,
                                                   float* __restrict__ z, int m) {
  __shared__ float ps[2 * HID];
  if (threadIdx.x < 2 * HID) ps[threadIdx.x] = pool[threadIdx.x];
  __syncthreads();
  for (int j = blockIdx.x * blockDim.x + threadIdx.x; j < m; j += gridDim.x * blockDim.x) {
    float acc = b[j];
#pragma unroll 8
    for (int k = 0; k < 2 * HID; ++k) acc += ps[k] * W[(size_t)k * m + j];
    if (conn && conn[j]) acc = MASK_VALUE;
    z[j] = (acc + g[j]) * (1.0f / TAU);
  }
}

// -------------------------------------------------------- max / argmax (first-occurrence)
__device__ __forceinline__ void amax_upd(float v, int ix, float& bv, int& bi) {
  if (v > bv || (v == bv && ix < bi)) { bv = v; bi = ix; }
}
__global__ __launch_bounds__(256) void k_max_stage1(const float* __restrict__ z, int n,
                                                    float* pmax, int* pidx) {
  float best = -INFINITY; int bidx = 0x7fffffff;
  for (int i = blockIdx.x * 256 + threadIdx.x; i < n; i += gridDim.x * 256)
    amax_upd(z[i], i, best, bidx);
  __shared__ float sm[256]; __shared__ int si[256];
  sm[threadIdx.x] = best; si[threadIdx.x] = bidx;
  __syncthreads();
  for (int s = 128; s > 0; s >>= 1) {
    if (threadIdx.x < s) amax_upd(sm[threadIdx.x + s], si[threadIdx.x + s],
                                  sm[threadIdx.x], si[threadIdx.x]);
    __syncthreads();
  }
  if (threadIdx.x == 0) { pmax[blockIdx.x] = sm[0]; pidx[blockIdx.x] = si[0]; }
}
__global__ __launch_bounds__(256) void k_max_stage2(const float* pmax, const int* pidx, int nb,
                                                    float* gmax, int* gidx) {
  float best = -INFINITY; int bidx = 0x7fffffff;
  for (int i = threadIdx.x; i < nb; i += 256) amax_upd(pmax[i], pidx[i], best, bidx);
  __shared__ float sm[256]; __shared__ int si[256];
  sm[threadIdx.x] = best; si[threadIdx.x] = bidx;
  __syncthreads();
  for (int s = 128; s > 0; s >>= 1) {
    if (threadIdx.x < s) amax_upd(sm[threadIdx.x + s], si[threadIdx.x + s],
                                  sm[threadIdx.x], si[threadIdx.x]);
    __syncthreads();
  }
  if (threadIdx.x == 0) { *gmax = sm[0]; *gidx = si[0]; }
}

// -------------------------------------------------------- exp + deterministic tree sum
__global__ __launch_bounds__(256) void k_expsum_stage1(float* __restrict__ z, int n,
                                                       const float* gmax, float* psum) {
  float m = *gmax;
  float acc = 0.0f;
  for (int i = blockIdx.x * 256 + threadIdx.x; i < n; i += gridDim.x * 256) {
    float e = __expf(z[i] - m);
    z[i] = e;
    acc += e;
  }
  __shared__ float sm[256];
  sm[threadIdx.x] = acc;
  __syncthreads();
  for (int s = 128; s > 0; s >>= 1) {
    if (threadIdx.x < s) sm[threadIdx.x] += sm[threadIdx.x + s];
    __syncthreads();
  }
  if (threadIdx.x == 0) psum[blockIdx.x] = sm[0];
}
__global__ __launch_bounds__(256) void k_sum_stage2(const float* psum, int nb, float* gsum) {
  float acc = 0.0f;
  for (int i = threadIdx.x; i < nb; i += 256) acc += psum[i];
  __shared__ float sm[256];
  sm[threadIdx.x] = acc;
  __syncthreads();
  for (int s = 128; s > 0; s >>= 1) {
    if (threadIdx.x < s) sm[threadIdx.x] += sm[threadIdx.x + s];
    __syncthreads();
  }
  if (threadIdx.x == 0) *gsum = sm[0];
}
__global__ void k_normalize(const float* __restrict__ z, int n, const float* gsum,
                            float* __restrict__ out) {
  float inv = 1.0f / *gsum;
  for (int i = blockIdx.x * blockDim.x + threadIdx.x; i < n; i += gridDim.x * blockDim.x)
    out[i] = z[i] * inv;
}

// -------------------------------------------------------- connectivity mask for head 2
__global__ void k_conn(const int* __restrict__ src, const int* __restrict__ dst, int e,
                       const int* i1p, int* __restrict__ conn) {
  int prev = *i1p;
  if (blockIdx.x == 0 && threadIdx.x == 0) conn[prev] = 1;
  for (int i = blockIdx.x * blockDim.x + threadIdx.x; i < e; i += gridDim.x * blockDim.x) {
    int s = src[i], d = dst[i];
    if (s == prev || d == prev) { conn[s] = 1; conn[d] = 1; }
  }
}

// -------------------------------------------------------- edge-type + stop heads
__global__ __launch_bounds__(32) void k_final(const float* __restrict__ gcn, const int* i1p,
                                              const int* i2p, const float* __restrict__ eW,
                                              const float* __restrict__ eb,
                                              const float* __restrict__ ge,
                                              float* __restrict__ out4) {
  __shared__ float ez[3];
  int t = threadIdx.x;
  int i1 = *i1p, i2 = *i2p;
  if (t < 3) {
    float acc = eb[t];
    const float* r1 = gcn + (size_t)i1 * HID;
    const float* r2 = gcn + (size_t)i2 * HID;
    for (int k = 0; k < HID; ++k) acc += r1[k] * eW[k * 3 + t];
    for (int k = 0; k < HID; ++k) acc += r2[k] * eW[(HID + k) * 3 + t];
    ez[t] = (acc + ge[t]) * (1.0f / TAU);
  }
  __syncthreads();
  if (t < 3) {
    float m = fmaxf(ez[0], fmaxf(ez[1], ez[2]));
    float s = __expf(ez[0] - m) + __expf(ez[1] - m) + __expf(ez[2] - m);
    out4[t] = __expf(ez[t] - m) / s;
  }
  if (t == 0) out4[3] = 1.0f;   // softmax over a single logit is identically 1
}

// ================================================================ launcher
extern "C" void kernel_launch(void* const* d_in, const int* in_sizes, int n_in,
                              void* d_out, int out_size, void* d_ws, size_t ws_size,
                              hipStream_t stream) {
  (void)in_sizes; (void)n_in; (void)out_size; (void)ws_size;

  const float* x    = (const float*)d_in[0];
  const int*   ei   = (const int*)d_in[1];
  const float* Ws[3] = {(const float*)d_in[2], (const float*)d_in[4], (const float*)d_in[6]};
  const float* bs[3] = {(const float*)d_in[3], (const float*)d_in[5], (const float*)d_in[7]};
  const float* n1W = (const float*)d_in[8];
  const float* n1b = (const float*)d_in[9];
  const float* n2W = (const float*)d_in[10];
  const float* n2b = (const float*)d_in[11];
  const float* eW  = (const float*)d_in[12];
  const float* eb  = (const float*)d_in[13];
  const float* g1  = (const float*)d_in[16];
  const float* g2  = (const float*)d_in[17];
  const float* ge  = (const float*)d_in[18];
  float* out = (float*)d_out;

  const int* src = ei;            // edge_index[0,:]
  const int* dst = ei + NEDGES;   // edge_index[1,:]

  // ---- workspace bump allocator (256B aligned)
  size_t off = 0;
  auto alloc = [&](size_t bytes) -> void* {
    off = (off + 255) & ~(size_t)255;
    void* p = (char*)d_ws + off;
    off += bytes;
    return p;
  };
  _Float16* actH = (_Float16*)alloc((size_t)NNODES * HID * 2);
  _Float16* WtH  = (_Float16*)alloc((size_t)HID * HID * 2);
  float* hbuf  = (float*)alloc((size_t)NNODES * HID * 4);
  float* outb  = (float*)alloc((size_t)NNODES * HID * 4);   // becomes gcn after layer 3
  float* dinv  = (float*)alloc((size_t)NNODES * 4);
  const int NBpool = 240;
  float* poolPS = (float*)alloc((size_t)NBpool * HID * 4);
  float* poolPM = (float*)alloc((size_t)NBpool * HID * 4);
  float* poolv  = (float*)alloc(2 * HID * 4);
  float* z      = (float*)alloc((size_t)MAXNN * 4);
  int*   conn   = (int*)alloc((size_t)MAXNN * 4);
  const int NBr = 512;
  float* pmax = (float*)alloc(NBr * 4);
  int*   pidx = (int*)alloc(NBr * 4);
  float* psum = (float*)alloc(NBr * 4);
  float* gmax = (float*)alloc(4);
  float* gsum = (float*)alloc(4);
  int*   gidx1 = (int*)alloc(4);
  int*   gidx2 = (int*)alloc(4);

  const int total = NNODES * HID;
  const int GS = 1024;                         // generic grid for grid-stride kernels

  // ---- degrees / symmetric norm
  k_zero_f32<<<GS, 256, 0, stream>>>(dinv, NNODES);
  k_deg<<<GS, 256, 0, stream>>>(dst, NEDGES, dinv);
  k_dinv<<<GS, 256, 0, stream>>>(dinv, NNODES);

  // ---- activations to f16
  k_f32_to_f16<<<GS, 256, 0, stream>>>(x, actH, total);

  // ---- three GCN layers (GEMM fuses bias + self-loop init of the scatter target)
  for (int l = 0; l < 3; ++l) {
    k_w_transpose_f16<<<(HID * HID + 255) / 256, 256, 0, stream>>>(Ws[l], WtH);
    k_gemm_f16<<<NNODES / 32, 256, 0, stream>>>(actH, WtH, bs[l], dinv, hbuf, outb); // WMMA
    k_edge_scatter<<<(NEDGES + 7) / 8, 256, 0, stream>>>(src, dst, hbuf, dinv, outb, NEDGES);
    k_relu_cvt<<<GS, 256, 0, stream>>>(outb, actH, total);
  }
  // outb now holds gcn = relu(layer3) in f32

  // ---- pool = [mean | max]
  k_pool_stage1<<<NBpool, 128, 0, stream>>>(outb, NNODES, poolPS, poolPM);
  k_pool_stage2<<<1, 128, 0, stream>>>(poolPS, poolPM, NBpool, NNODES, poolv);

  // ---- head 1: z1 -> softmax -> out[0:MAXN]; argmax -> i1
  k_head_gemv<<<GS, 256, 0, stream>>>(poolv, n1W, n1b, g1, nullptr, z, MAXNN);
  k_max_stage1<<<NBr, 256, 0, stream>>>(z, MAXNN, pmax, pidx);
  k_max_stage2<<<1, 256, 0, stream>>>(pmax, pidx, NBr, gmax, gidx1);
  k_expsum_stage1<<<NBr, 256, 0, stream>>>(z, MAXNN, gmax, psum);
  k_sum_stage2<<<1, 256, 0, stream>>>(psum, NBr, gsum);
  k_normalize<<<GS, 256, 0, stream>>>(z, MAXNN, gsum, out);

  // ---- connectivity mask around i1
  k_zero_i32<<<GS, 256, 0, stream>>>(conn, MAXNN);
  k_conn<<<GS, 256, 0, stream>>>(src, dst, NEDGES, gidx1, conn);

  // ---- head 2: masked logits -> softmax -> out[MAXN:2*MAXN]; argmax -> i2
  k_head_gemv<<<GS, 256, 0, stream>>>(poolv, n2W, n2b, g2, conn, z, MAXNN);
  k_max_stage1<<<NBr, 256, 0, stream>>>(z, MAXNN, pmax, pidx);
  k_max_stage2<<<1, 256, 0, stream>>>(pmax, pidx, NBr, gmax, gidx2);
  k_expsum_stage1<<<NBr, 256, 0, stream>>>(z, MAXNN, gmax, psum);
  k_sum_stage2<<<1, 256, 0, stream>>>(psum, NBr, gsum);
  k_normalize<<<GS, 256, 0, stream>>>(z, MAXNN, gsum, out + MAXNN);

  // ---- edge-type softmax (3) + stop head (1)
  k_final<<<1, 32, 0, stream>>>(outb, gidx1, gidx2, eW, eb, ge, out + 2 * MAXNN);
}
